// MHCSA_5884105195619
// MI455X (gfx1250) — matrix operand order repaired and checked by reading, not compiled, not run on hardware
//
#include <hip/hip_runtime.h>
#include <hip/hip_bf16.h>

// Problem constants (B,C,H,W) = (8,64,64,64)
#define BATCH 8
#define CDIM  64
#define HDIM  64
#define WDIM  64
#define NTOK  (HDIM*WDIM)   // 4096
#define FWAVES 4            // independent waves per flash block

typedef __bf16 bf16_t;
typedef __attribute__((ext_vector_type(8)))  __bf16 v8bf;
typedef __attribute__((ext_vector_type(16))) __bf16 v16bf;
typedef __attribute__((ext_vector_type(8)))  float  v8f;

// ---------------------------------------------------------------------------
// WMMA helpers (CDNA5 v_wmma_f32_16x16x32_bf16)
// ---------------------------------------------------------------------------
__device__ __forceinline__ v8f wmma_bf16(v16bf a, v16bf b, v8f c) {
  // 8 args: (neg_a, A, neg_b, B, c_mod, C, reuse_a, reuse_b)
  return __builtin_amdgcn_wmma_f32_16x16x32_bf16(false, a, false, b, (short)0, c,
                                                 false, false);
}

// A-matrix 16x32 bf16 fragment, this code's contiguous K order (not the ISA table order):
//   lane half = lane/16; element e -> K = half*16+e (contiguous, matches B loads)
// rowbase points at K=0 of this lane's row; kofs selects the 32-wide K block.
__device__ __forceinline__ v16bf load_a_frag(const bf16_t* rowbase, int half, int kofs) {
  v8bf lo = *(const v8bf*)(rowbase + kofs + half * 16);
  v8bf hi = *(const v8bf*)(rowbase + kofs + half * 16 + 8);
  return __builtin_shufflevector(lo, hi, 0, 1, 2, 3, 4, 5, 6, 7,
                                 8, 9, 10, 11, 12, 13, 14, 15);
}

// ---------------------------------------------------------------------------
// Fusion-attention small kernels
// ---------------------------------------------------------------------------

// Global avg/max pool over H*W per (b,c): one block per (b*C+c)
__global__ void pool_kernel(const float* __restrict__ X2,
                            float* __restrict__ pavg, float* __restrict__ pmax) {
  __shared__ float ssum[256];
  __shared__ float smax[256];
  const int bc = blockIdx.x;
  const float* p = X2 + (size_t)bc * NTOK;
  float s = 0.f, m = -1e30f;
  for (int i = threadIdx.x; i < NTOK; i += 256) {
    float v = p[i];
    s += v;
    m = fmaxf(m, v);
  }
  ssum[threadIdx.x] = s;
  smax[threadIdx.x] = m;
  __syncthreads();
  for (int st = 128; st > 0; st >>= 1) {
    if (threadIdx.x < st) {
      ssum[threadIdx.x] += ssum[threadIdx.x + st];
      smax[threadIdx.x] = fmaxf(smax[threadIdx.x], smax[threadIdx.x + st]);
    }
    __syncthreads();
  }
  if (threadIdx.x == 0) {
    pavg[bc] = ssum[0] * (1.0f / NTOK);
    pmax[bc] = smax[0];
  }
}

// Per-pixel mean/max over channels -> mc/mx [B,N]
__global__ void reduce_c_kernel(const float* __restrict__ X2,
                                float* __restrict__ mc, float* __restrict__ mx) {
  const int idx = blockIdx.x * blockDim.x + threadIdx.x;  // b*N + n
  const int b = idx / NTOK, n = idx % NTOK;
  const float* p = X2 + (size_t)b * CDIM * NTOK + n;
  float s = 0.f, m = -1e30f;
  for (int c = 0; c < CDIM; ++c) {
    float v = p[(size_t)c * NTOK];
    s += v;
    m = fmaxf(m, v);
  }
  mc[idx] = s * (1.0f / CDIM);
  mx[idx] = m;
}

// 7x7 spatial conv (2 input planes), zero pad 3 -> sp [B,N]
__global__ void spatial_kernel(const float* __restrict__ mc, const float* __restrict__ mx,
                               const float* __restrict__ sw, const float* __restrict__ sb,
                               float* __restrict__ sp) {
  const int idx = blockIdx.x * blockDim.x + threadIdx.x;  // b*N + n
  const int b = idx / NTOK, n = idx % NTOK;
  const int h = n >> 6, w = n & 63;
  const float* p0 = mc + (size_t)b * NTOK;
  const float* p1 = mx + (size_t)b * NTOK;
  float acc = sb[0];
  for (int dy = 0; dy < 7; ++dy) {
    int y = h + dy - 3;
    if (y < 0 || y >= HDIM) continue;
    for (int dx = 0; dx < 7; ++dx) {
      int xw = w + dx - 3;
      if (xw < 0 || xw >= WDIM) continue;
      int nn = y * WDIM + xw;
      acc += p0[nn] * sw[dy * 7 + dx] + p1[nn] * sw[49 + dy * 7 + dx];
    }
  }
  sp[idx] = acc;
}

// Channel branch conv1d(k=5,pad=2) then project:
//   cterm[b,o] = bck[o] + sum_c Wck[o,c]*ch[b,c];  wsum[o] = sum_c Wck[o,c]
__global__ void channel_kernel(const float* __restrict__ pavg, const float* __restrict__ pmax,
                               const float* __restrict__ cw, const float* __restrict__ Wck,
                               const float* __restrict__ bck, float* __restrict__ cterm,
                               float* __restrict__ wsum) {
  __shared__ float chs[CDIM];
  const int b = blockIdx.x, t = threadIdx.x;  // 64 threads
  {
    float acc = 0.f;
    for (int d = 0; d < 5; ++d) {
      int c = t + d - 2;
      if (c >= 0 && c < CDIM)
        acc += pavg[b * CDIM + c] * cw[d] + pmax[b * CDIM + c] * cw[5 + d];
    }
    chs[t] = acc;
  }
  __syncthreads();
  float ct = bck[t], ws = 0.f;
  for (int c = 0; c < CDIM; ++c) {
    float wv = Wck[t * CDIM + c];
    ct += wv * chs[c];
    ws += wv;
  }
  cterm[b * CDIM + t] = ct;
  if (b == 0) wsum[t] = ws;
}

// ---------------------------------------------------------------------------
// 1x1-conv projection GEMM: out[o,n] = bias[o] + sum_c W[o,c]*x[c,n]
// stores bf16 transposed ([N,C]) and/or natural ([C,N]) layouts
// ---------------------------------------------------------------------------
__global__ void proj_kernel(const float* __restrict__ W, const float* __restrict__ bias,
                            const float* __restrict__ X, bf16_t* __restrict__ outT,
                            bf16_t* __restrict__ outN) {
  const int n = blockIdx.x * blockDim.x + threadIdx.x;
  const int b = blockIdx.y;
  const float* xb = X + (size_t)b * CDIM * NTOK;
  float xc[CDIM];
#pragma unroll
  for (int c = 0; c < CDIM; ++c) xc[c] = xb[(size_t)c * NTOK + n];
  for (int o = 0; o < CDIM; ++o) {
    float acc = bias[o];
#pragma unroll
    for (int c = 0; c < CDIM; ++c) acc += W[o * CDIM + c] * xc[c];
    if (outT) outT[((size_t)b * NTOK + n) * CDIM + o] = (bf16_t)acc;
    if (outN) outN[(size_t)b * CDIM * NTOK + (size_t)o * NTOK + n] = (bf16_t)acc;
  }
}

// position-bias vectors: a[m] = cterm . Q[:,m] ; g[m] = wsum . Q[:,m]
__global__ void ag_kernel(const bf16_t* __restrict__ QT, const float* __restrict__ cterm,
                          const float* __restrict__ wsum, float* __restrict__ aW,
                          float* __restrict__ gW) {
  const int idx = blockIdx.x * blockDim.x + threadIdx.x;  // b*N + m
  const int b = idx / NTOK;
  const bf16_t* q = QT + (size_t)idx * CDIM;
  const float* ct = cterm + b * CDIM;
  float a = 0.f, g = 0.f;
#pragma unroll
  for (int o = 0; o < CDIM; ++o) {
    float qv = (float)q[o];
    a += ct[o] * qv;
    g += wsum[o] * qv;
  }
  aW[idx] = a;
  gW[idx] = g;
}

// ---------------------------------------------------------------------------
// Flash attention core (WMMA bf16, online softmax, never materializes [N,N])
//   S[n,m] = Q^T K + a[m] + sp[n]*g[m];  O^T[n,c] = softmax_m(S) . V^T
// One wave owns 16 rows; iterates m in steps of 32.
// ---------------------------------------------------------------------------
__global__ void __launch_bounds__(FWAVES * 32) flash_kernel(
    const bf16_t* __restrict__ QT, const bf16_t* __restrict__ KT,
    const bf16_t* __restrict__ Vb, const float* __restrict__ aW,
    const float* __restrict__ gW, const float* __restrict__ spW,
    float* __restrict__ outW) {
  __shared__ __align__(64) bf16_t ptile[FWAVES][16 * 32];  // exp(S) staging per wave
  const int tid = threadIdx.x;
  const int wave = tid >> 5, lane = tid & 31;
  const int half = lane >> 4, l16 = lane & 15;
  const int gw = blockIdx.x * FWAVES + wave;
  const int b = gw / (NTOK / 16);
  const int n0 = (gw % (NTOK / 16)) * 16;

  const bf16_t* qt = QT + (size_t)b * NTOK * CDIM;
  const bf16_t* kt = KT + (size_t)b * NTOK * CDIM;
  const bf16_t* vb = Vb + (size_t)b * CDIM * NTOK;
  const float* av = aW + (size_t)b * NTOK;
  const float* gv = gW + (size_t)b * NTOK;
  const float* sv = spW + (size_t)b * NTOK;
  float* ob = outW + (size_t)b * CDIM * NTOK;

  // persistent Q A-fragments (c = 0..31 and 32..63)
  const bf16_t* qrow = qt + (size_t)(n0 + l16) * CDIM;
  v16bf qa0 = load_a_frag(qrow, half, 0);
  v16bf qa1 = load_a_frag(qrow, half, 32);

  float sp_r[8];
#pragma unroll
  for (int r = 0; r < 8; ++r) sp_r[r] = sv[n0 + half * 8 + r];

  float m_run[8], lsum[8];
#pragma unroll
  for (int r = 0; r < 8; ++r) {
    m_run[r] = -1e30f;
    lsum[r] = 0.f;
  }
  v8f o0 = {}, o1 = {}, o2 = {}, o3 = {};
  bf16_t* pt = &ptile[wave][0];

  for (int m0 = 0; m0 < NTOK; m0 += 32) {
    // ---- score sub-tiles (D layout: M=n rows, N=m cols) ----
    v8f s0 = {}, s1 = {};
    {
      const bf16_t* krow0 = kt + (size_t)(m0 + l16) * CDIM;
      const bf16_t* krow1 = kt + (size_t)(m0 + 16 + l16) * CDIM;
      v16bf bk;
      bk = *(const v16bf*)(krow0 + half * 16);       s0 = wmma_bf16(qa0, bk, s0);
      bk = *(const v16bf*)(krow0 + 32 + half * 16);  s0 = wmma_bf16(qa1, bk, s0);
      bk = *(const v16bf*)(krow1 + half * 16);       s1 = wmma_bf16(qa0, bk, s1);
      bk = *(const v16bf*)(krow1 + 32 + half * 16);  s1 = wmma_bf16(qa1, bk, s1);
    }
    if (m0 + 32 < NTOK)
      __builtin_prefetch(kt + (size_t)(m0 + 32 + l16) * CDIM, 0, 1);

    // position bias: S += a[m] + sp[n]*g[m]
    const float a0 = av[m0 + l16], g0 = gv[m0 + l16];
    const float a1_ = av[m0 + 16 + l16], g1_ = gv[m0 + 16 + l16];
    float tmax[8];
#pragma unroll
    for (int r = 0; r < 8; ++r) {
      s0[r] += a0 + sp_r[r] * g0;
      s1[r] += a1_ + sp_r[r] * g1_;
      tmax[r] = fmaxf(s0[r], s1[r]);
    }
    // row max across the 16 lanes of this half (xor 1,2,4,8 stays in-half)
#pragma unroll
    for (int r = 0; r < 8; ++r) {
#pragma unroll
      for (int off = 1; off < 16; off <<= 1)
        tmax[r] = fmaxf(tmax[r], __shfl_xor(tmax[r], off, 32));
    }
    // online softmax update + exp tile to LDS (bf16)
#pragma unroll
    for (int r = 0; r < 8; ++r) {
      float nm = fmaxf(m_run[r], tmax[r]);
      float corr = __expf(m_run[r] - nm);
      m_run[r] = nm;
      lsum[r] *= corr;
      o0[r] *= corr; o1[r] *= corr; o2[r] *= corr; o3[r] *= corr;
      float p0 = __expf(s0[r] - nm);
      float p1 = __expf(s1[r] - nm);
      lsum[r] += p0 + p1;  // per-lane partial row sum (reduced at end)
      pt[(half * 8 + r) * 32 + l16] = (bf16_t)p0;
      pt[(half * 8 + r) * 32 + 16 + l16] = (bf16_t)p1;
    }
    __builtin_amdgcn_wave_barrier();  // LDS per-wave is in-order; stop reordering
    v16bf pa = load_a_frag(pt + l16 * 32, half, 0);  // P' [16n x 32m] A-fragment
    __builtin_amdgcn_wave_barrier();

    // O^T[n,c] += P' x V^T : B fragments contiguous along m in V[c,:]
    {
      const bf16_t* vbase = vb + m0 + half * 16;
      v16bf b0 = *(const v16bf*)(vbase + (size_t)(0 * 16 + l16) * NTOK);
      v16bf b1 = *(const v16bf*)(vbase + (size_t)(1 * 16 + l16) * NTOK);
      v16bf b2 = *(const v16bf*)(vbase + (size_t)(2 * 16 + l16) * NTOK);
      v16bf b3 = *(const v16bf*)(vbase + (size_t)(3 * 16 + l16) * NTOK);
      o0 = wmma_bf16(pa, b0, o0);
      o1 = wmma_bf16(pa, b1, o1);
      o2 = wmma_bf16(pa, b2, o2);
      o3 = wmma_bf16(pa, b3, o3);
    }
  }

  // finalize: reduce row sums across lanes, normalize, store out[c,n] fp32
#pragma unroll
  for (int r = 0; r < 8; ++r) {
#pragma unroll
    for (int off = 1; off < 16; off <<= 1) lsum[r] += __shfl_xor(lsum[r], off, 32);
    float inv = 1.0f / lsum[r];
    o0[r] *= inv; o1[r] *= inv; o2[r] *= inv; o3[r] *= inv;
  }
#pragma unroll
  for (int r = 0; r < 8; ++r) {
    int n = n0 + half * 8 + r;
    ob[(size_t)(0 * 16 + l16) * NTOK + n] = o0[r];
    ob[(size_t)(1 * 16 + l16) * NTOK + n] = o1[r];
    ob[(size_t)(2 * 16 + l16) * NTOK + n] = o2[r];
    ob[(size_t)(3 * 16 + l16) * NTOK + n] = o3[r];
  }
}

// ---------------------------------------------------------------------------
// Epilogue: attn1/attn2 1x1 convs + 2-way softmax blend of [x, x2]
// ---------------------------------------------------------------------------
__global__ void epilogue_kernel(const float* __restrict__ outW,
                                const float* __restrict__ Wc1, const float* __restrict__ bc1,
                                const float* __restrict__ Wc2, const float* __restrict__ bc2,
                                const float* __restrict__ X, const float* __restrict__ X2,
                                float* __restrict__ Y) {
  const int idx = blockIdx.x * blockDim.x + threadIdx.x;  // b*N + n
  const int b = idx / NTOK, n = idx % NTOK;
  const float* ocol = outW + (size_t)b * CDIM * NTOK + n;
  float oc[CDIM];
#pragma unroll
  for (int c = 0; c < CDIM; ++c) oc[c] = ocol[(size_t)c * NTOK];
  for (int co = 0; co < CDIM; ++co) {
    float a1 = bc1[co], a2 = bc2[co];
#pragma unroll
    for (int c = 0; c < CDIM; ++c) {
      a1 += Wc1[co * CDIM + c] * oc[c];
      a2 += Wc2[co * CDIM + c] * oc[c];
    }
    float mv = fmaxf(a1, a2);
    float e1 = __expf(a1 - mv), e2 = __expf(a2 - mv);
    size_t gi = (size_t)b * CDIM * NTOK + (size_t)co * NTOK + n;
    Y[gi] = (X[gi] * e1 + X2[gi] * e2) / (e1 + e2);
  }
}

// ---------------------------------------------------------------------------
extern "C" void kernel_launch(void* const* d_in, const int* in_sizes, int n_in,
                              void* d_out, int out_size, void* d_ws, size_t ws_size,
                              hipStream_t stream) {
  (void)in_sizes; (void)n_in; (void)out_size; (void)ws_size;
  const float* x     = (const float*)d_in[0];
  const float* x2    = (const float*)d_in[1];
  const float* Wq    = (const float*)d_in[2];
  const float* bq    = (const float*)d_in[3];
  const float* Wk    = (const float*)d_in[4];
  const float* bk    = (const float*)d_in[5];
  const float* Wv    = (const float*)d_in[6];
  const float* bv    = (const float*)d_in[7];
  const float* Wck   = (const float*)d_in[8];
  const float* bck   = (const float*)d_in[9];
  const float* fa_cw = (const float*)d_in[10];
  const float* fa_sw = (const float*)d_in[11];
  const float* fa_sb = (const float*)d_in[12];
  const float* Wc1   = (const float*)d_in[13];
  const float* bc1   = (const float*)d_in[14];
  const float* Wc2   = (const float*)d_in[15];
  const float* bc2   = (const float*)d_in[16];
  float* Y = (float*)d_out;

  // workspace carve-up (256B aligned)
  char* ws = (char*)d_ws;
  size_t off = 0;
  auto carve = [&](size_t bytes) {
    char* p = ws + off;
    off = (off + bytes + 255) & ~(size_t)255;
    return p;
  };
  float* spW   = (float*)carve((size_t)BATCH * NTOK * 4);
  float* aW    = (float*)carve((size_t)BATCH * NTOK * 4);
  float* gW    = (float*)carve((size_t)BATCH * NTOK * 4);
  float* mcW   = (float*)carve((size_t)BATCH * NTOK * 4);
  float* mxW   = (float*)carve((size_t)BATCH * NTOK * 4);
  float* pavg  = (float*)carve((size_t)BATCH * CDIM * 4);
  float* pmax  = (float*)carve((size_t)BATCH * CDIM * 4);
  float* cterm = (float*)carve((size_t)BATCH * CDIM * 4);
  float* wsum  = (float*)carve((size_t)CDIM * 4);
  bf16_t* QT   = (bf16_t*)carve((size_t)BATCH * NTOK * CDIM * 2);
  bf16_t* KT   = (bf16_t*)carve((size_t)BATCH * NTOK * CDIM * 2);
  bf16_t* Vb   = (bf16_t*)carve((size_t)BATCH * CDIM * NTOK * 2);
  float* outW  = (float*)carve((size_t)BATCH * CDIM * NTOK * 4);

  const int pix_blocks = BATCH * NTOK / 256;  // 128

  // fusion-attention branch
  pool_kernel<<<BATCH * CDIM, 256, 0, stream>>>(x2, pavg, pmax);
  reduce_c_kernel<<<pix_blocks, 256, 0, stream>>>(x2, mcW, mxW);
  spatial_kernel<<<pix_blocks, 256, 0, stream>>>(mcW, mxW, fa_sw, fa_sb, spW);
  channel_kernel<<<BATCH, CDIM, 0, stream>>>(pavg, pmax, fa_cw, Wck, bck, cterm, wsum);

  // projections (bf16, WMMA-friendly layouts)
  dim3 pg(NTOK / 256, BATCH);
  proj_kernel<<<pg, 256, 0, stream>>>(Wq, bq, x, QT, nullptr);
  proj_kernel<<<pg, 256, 0, stream>>>(Wk, bk, x, KT, nullptr);
  proj_kernel<<<pg, 256, 0, stream>>>(Wv, bv, x, nullptr, Vb);

  // rank-1 position-bias vectors
  ag_kernel<<<pix_blocks, 256, 0, stream>>>(QT, cterm, wsum, aW, gW);

  // fused flash attention (WMMA core)
  const int fblocks = BATCH * (NTOK / 16) / FWAVES;  // 512
  flash_kernel<<<fblocks, FWAVES * 32, 0, stream>>>(QT, KT, Vb, aW, gW, spW, outW);

  // epilogue
  epilogue_kernel<<<pix_blocks, 256, 0, stream>>>(outW, Wc1, bc1, Wc2, bc2, x, x2, Y);
}
